// Attention_9036611191301
// MI455X (gfx1250) — matrix operand order repaired and checked by reading, not compiled, not run on hardware
//
#include <hip/hip_runtime.h>

// ---------------------------------------------------------------------------
// Flash attention for gfx1250 (MI455X): bf16 WMMA, fp32 accumulate/softmax.
// B=16, Q=K=2048, D=512.  BQ=64 queries/WG, 512 threads (16 waves):
// wave = (qsub 0..3) x (kq/dslice 0..3).  K and V staged in LDS as bf16 to
// cut L2 traffic 4x vs per-wave global fetch (L2-bandwidth roofline).
// ---------------------------------------------------------------------------

typedef __attribute__((ext_vector_type(16))) __bf16 v16bf;
typedef __attribute__((ext_vector_type(4)))  __bf16 v4bf;
typedef __attribute__((ext_vector_type(8)))  float  v8f;
typedef __attribute__((ext_vector_type(4)))  float  v4f;
typedef __attribute__((ext_vector_type(4)))  int    v4i;

#define B_SZ 16
#define QN   2048
#define KN   2048
#define DN   512
#define BQ   64      // query rows per workgroup
#define BK   64      // keys per iteration
#define QSTR 520     // Q LDS row stride (bf16), 16B-aligned + bank skew
#define KSTR 520     // K LDS row stride (bf16)
#define VSTR 72      // V^T LDS row stride (bf16): vlds[col][key]
#define SSTR 68      // score LDS row stride (f32)
#define PSTR 72      // P LDS row stride (bf16)
#define NEG_INF_F (-1e30f)

__device__ __forceinline__ float lane_bcast_f(float v, int srcLane) {
  return __int_as_float(__builtin_amdgcn_ds_bpermute(srcLane << 2, __float_as_int(v)));
}

__device__ __forceinline__ v4bf cvt4(v4f f) {
  v4bf r;
  r[0] = (__bf16)f[0]; r[1] = (__bf16)f[1];
  r[2] = (__bf16)f[2]; r[3] = (__bf16)f[3];
  return r;
}

// A-matrix 16x32 bf16 fragment: two 16B runs 16 bf16 apart (cols c, c+16)
__device__ __forceinline__ v16bf load_a_frag(const __bf16* p) {
  union { v4i q[2]; v16bf v; } u;
  u.q[0] = *(const v4i*)(p);
  u.q[1] = *(const v4i*)(p + 16);
  return u.v;
}

// B-matrix 32x16 bf16 fragment: two 16B runs 16 bf16 apart (k, k+16)
__device__ __forceinline__ v16bf load_b_frag(const __bf16* p) {
  union { v4i q[2]; v16bf v; } u;
  u.q[0] = *(const v4i*)(p);
  u.q[1] = *(const v4i*)(p + 16);
  return u.v;
}

__global__ __launch_bounds__(512)
void attn_flash_bf16(const float* __restrict__ Qm, const float* __restrict__ Km,
                     const float* __restrict__ Vm, const int* __restrict__ Mm,
                     float* __restrict__ Om) {
  __shared__ __bf16 qlds[BQ * QSTR];    // Q tile  64x512 bf16  (66.5 KB)
  __shared__ __bf16 klds[BK * KSTR];    // K tile  64x512 bf16  (66.5 KB)
  __shared__ __bf16 vlds[DN * VSTR];    // V tile transposed [col][key] (73.7 KB)
  __shared__ float  slds[BQ * SSTR];    // scores 64x64 f32     (17.4 KB)
  __shared__ __bf16 plds[BQ * PSTR];    // P      64x64 bf16    ( 9.2 KB)

  const int tid  = threadIdx.x;
  const int lane = tid & 31;
  const int wave = tid >> 5;            // 0..15
  const int qsub = wave >> 2;           // 0..3 : which 16 query rows
  const int kq   = wave & 3;            // 0..3 : key sub-tile (stage1) / D-slice (stage3)
  const int hi   = lane >> 4;
  const int ln   = lane & 15;

  const int qb = blockIdx.x % (QN / BQ);
  const int b  = blockIdx.x / (QN / BQ);
  const int q0 = qb * BQ;

  const float scale = 0.044194173824159216f;  // 1/sqrt(512)

  const float* kbase = Km + (size_t)b * KN * DN;
  const float* vbase = Vm + (size_t)b * KN * DN;
  const int*   mbase = Mm + ((size_t)b * QN + q0) * KN;

  // ---- stage Q tile (64 x 512) -> bf16 LDS, coalesced b128 reads ----
  {
    const float* qsrc = Qm + ((size_t)b * QN + q0) * DN;
    for (int it = 0; it < 16; ++it) {
      int flat = (it * 512 + tid) * 4;
      int r = flat >> 9, c = flat & 511;
      v4f f = *(const v4f*)(qsrc + (size_t)r * DN + c);
      *(v4bf*)(qlds + r * QSTR + c) = cvt4(f);
    }
  }

  // O accumulator: 16 rows x 128 cols (this wave's slice) = 8 f32 tiles
  v8f acc[8];
#pragma unroll
  for (int t = 0; t < 8; ++t)
    acc[t] = (v8f){0.f, 0.f, 0.f, 0.f, 0.f, 0.f, 0.f, 0.f};

  float m_row = -3.0e38f;   // running max for row qsub*16 + ln
  float l_row = 0.0f;       // running sum

  for (int kb = 0; kb < KN; kb += BK) {
    // ============ stage 0: stage K (row-major) and V^T into LDS ============
    {
      const float* ksrc = kbase + (size_t)kb * DN;
#pragma unroll
      for (int it = 0; it < 16; ++it) {
        int flat = (it * 512 + tid) * 4;
        int r = flat >> 9, c = flat & 511;
        v4f f = *(const v4f*)(ksrc + (size_t)r * DN + c);
        *(v4bf*)(klds + r * KSTR + c) = cvt4(f);
      }
      const float* vsrc = vbase + (size_t)kb * DN;
      const int col = tid;  // blockDim == 512 == DN
#pragma unroll
      for (int kg = 0; kg < 16; ++kg) {
        int k0 = kg * 4;
        v4f f;
        f[0] = vsrc[(size_t)(k0 + 0) * DN + col];
        f[1] = vsrc[(size_t)(k0 + 1) * DN + col];
        f[2] = vsrc[(size_t)(k0 + 2) * DN + col];
        f[3] = vsrc[(size_t)(k0 + 3) * DN + col];
        *(v4bf*)(vlds + col * VSTR + k0) = cvt4(f);
      }
    }
    __syncthreads();

    // ============ stage 1: S tile = Q(qsub) * K(kq)^T, full-D reduction ====
    const int kt = kb + kq * 16;
    v8f s = (v8f){0.f, 0.f, 0.f, 0.f, 0.f, 0.f, 0.f, 0.f};
#pragma unroll
    for (int dk = 0; dk < DN; dk += 32) {
      v16bf a  = load_a_frag(qlds + (qsub * 16 + ln) * QSTR + dk + hi * 8);
      v16bf bb = load_b_frag(klds + (kq * 16 + ln) * KSTR + dk + hi * 8);
      s = __builtin_amdgcn_wmma_f32_16x16x32_bf16(false, a, false, bb,
                                                  (short)0, s, false, false);
    }
    // scale + mask in C-layout, share via LDS
#pragma unroll
    for (int r = 0; r < 8; ++r) {
      int mr = r + hi * 8;
      int mk = mbase[(size_t)(qsub * 16 + mr) * KN + kt + ln];
      float sv = s[r] * scale;
      sv = (mk == 0) ? NEG_INF_F : sv;
      slds[(qsub * 16 + mr) * SSTR + kq * 16 + ln] = sv;
    }
    __syncthreads();

    // ============ stage 2: online softmax (lane = row qsub*16+ln, half hi) =
    const float* srow = slds + (qsub * 16 + ln) * SSTR + hi * 32;
    float lm = -3.0e38f;
#pragma unroll
    for (int c = 0; c < 32; ++c) lm = fmaxf(lm, srow[c]);
    lm = fmaxf(lm, lane_bcast_f(lm, lane ^ 16));
    float m_new = fmaxf(m_row, lm);
    float alpha = __expf(m_row - m_new);

    float lsum = 0.0f;
    __bf16* prow = plds + (qsub * 16 + ln) * PSTR + hi * 32;
#pragma unroll
    for (int c = 0; c < 32; ++c) {
      float p = __expf(srow[c] - m_new);
      lsum += p;
      if (((hi * 32 + c) >> 4) == kq)   // the 4 waves sharing qsub split P cols
        prow[c] = (__bf16)p;
    }
    lsum += lane_bcast_f(lsum, lane ^ 16);
    l_row = l_row * alpha + lsum;
    m_row = m_new;

    // rescale accumulator tiles by per-row alpha (rows hi*8 + r of this qsub)
#pragma unroll
    for (int r = 0; r < 8; ++r) {
      float ar = lane_bcast_f(alpha, hi * 8 + r);
#pragma unroll
      for (int t = 0; t < 8; ++t) acc[t][r] *= ar;
    }
    __syncthreads();

    // ============ stage 3: O(qsub, dslice=kq) += P * V ======================
#pragma unroll
    for (int ks = 0; ks < 2; ++ks) {
      v16bf a = load_a_frag(plds + (qsub * 16 + ln) * PSTR + ks * 32 + hi * 8);
#pragma unroll
      for (int t = 0; t < 8; ++t) {
        int col = kq * 128 + t * 16 + ln;
        v16bf bb = load_b_frag(vlds + col * VSTR + ks * 32 + hi * 8);
        acc[t] = __builtin_amdgcn_wmma_f32_16x16x32_bf16(false, a, false, bb,
                                                         (short)0, acc[t],
                                                         false, false);
      }
    }
    __syncthreads();   // protect klds/vlds/plds before next block's staging
  }

  // ================= epilogue: normalize rows, store fp32 ==================
  float invl = 1.0f / l_row;
  float* orow = Om + ((size_t)b * QN + q0 + qsub * 16) * DN + kq * 128 + ln;
#pragma unroll
  for (int r = 0; r < 8; ++r) {
    int mr = r + hi * 8;
    float inv = lane_bcast_f(invl, hi * 8 + r);
#pragma unroll
    for (int t = 0; t < 8; ++t)
      orow[(size_t)mr * DN + t * 16] = acc[t][r] * inv;
  }
}

extern "C" void kernel_launch(void* const* d_in, const int* in_sizes, int n_in,
                              void* d_out, int out_size, void* d_ws, size_t ws_size,
                              hipStream_t stream) {
  const float* Q = (const float*)d_in[0];
  const float* K = (const float*)d_in[1];
  const float* V = (const float*)d_in[2];
  const int*   M = (const int*)d_in[3];
  float* O = (float*)d_out;
  dim3 grid(B_SZ * (QN / BQ));   // 512 workgroups
  dim3 block(512);               // 16 waves (wave32)
  hipLaunchKernelGGL(attn_flash_bf16, grid, block, 0, stream, Q, K, V, M, O);
}
